// GraphNN_40209483825155
// MI455X (gfx1250) — matrix operand order repaired and checked by reading, not compiled, not run on hardware
//
#include <hip/hip_runtime.h>

// GraphConv x3 on MI455X (gfx1250).
// Roofline: dense GEMMs are ~4 GFLOP (negligible); cost is L2-resident
// gather/scatter-add over 1.6M edges (all feature arrays fit in 192MB L2).
// Layer 3 (D_OUT=1) is rewritten as per-node scalar dot + 1-float/edge
// scatter (64x fewer atomics). Node transforms use v_wmma_f32_16x16x32_bf16.
// Atomics use unsafeAtomicAdd -> native GLOBAL_ATOMIC_ADD_F32 (no CAS loop).

typedef __attribute__((ext_vector_type(16))) __bf16 v16bf;
typedef __attribute__((ext_vector_type(8)))  float  v8f;

__device__ __forceinline__ void atom_add_f32(float* p, float v) {
  unsafeAtomicAdd(p, v);   // hardware global_atomic_add_f32
}

// ---------------------------------------------------------------- zero fill
__global__ void zero_f32_kernel(float* __restrict__ p, long long n) {
  long long i = (long long)blockIdx.x * blockDim.x + threadIdx.x;
  if (i < n) p[i] = 0.0f;
}

// ------------------------------------------------- edge scatter-add (K=2^LOGK)
// One thread per (edge, feature). Lanes of a wave cover consecutive features
// of the same edge -> coalesced 128B gather + single-line atomic burst.
template <int LOGK>
__global__ void scatter_add_kernel(const float* __restrict__ feat,
                                   const int* __restrict__ src,
                                   const int* __restrict__ dst,
                                   float* __restrict__ agg,
                                   long long total) {
  long long tid = (long long)blockIdx.x * blockDim.x + threadIdx.x;
  if (tid >= total) return;
  const int K = 1 << LOGK;
  long long e = tid >> LOGK;
  int f = (int)(tid & (K - 1));
  int s = src[e];
  int d = dst[e];
  float v = feat[(long long)s * K + f];
  atom_add_f32(&agg[(long long)d * K + f], v);
}

// --------------------------------------------- WMMA node transform (K=32/64)
// out[i,:] = act( agg[i,:]@w_rel + xin[i,:]@w_root + bias )
// Block = 16 nodes; 4 waves, wave w owns output cols [16w,16w+16).
__device__ __forceinline__ void pack8(v16bf& v, int base, float4 lo, float4 hi) {
  v[base + 0] = (__bf16)lo.x; v[base + 1] = (__bf16)lo.y;
  v[base + 2] = (__bf16)lo.z; v[base + 3] = (__bf16)lo.w;
  v[base + 4] = (__bf16)hi.x; v[base + 5] = (__bf16)hi.y;
  v[base + 6] = (__bf16)hi.z; v[base + 7] = (__bf16)hi.w;
}

template <int K, bool RELU>
__global__ void __launch_bounds__(128)
gconv_transform_kernel(const float* __restrict__ agg,
                       const float* __restrict__ xin,
                       const float* __restrict__ w_rel,   // [K,64] row-major
                       const float* __restrict__ w_root,  // [K,64] row-major
                       const float* __restrict__ bias,    // [64]
                       float* __restrict__ out,           // [N,64]
                       long long nNodes) {
  const int lane = threadIdx.x & 31;
  const int wave = threadIdx.x >> 5;       // 0..3 -> output col tile
  const int half = lane >> 4;              // lane group (0: lanes 0-15)
  const int idx  = lane & 15;
  const long long tileM = (long long)blockIdx.x * 16;
  const int n = wave * 16 + idx;           // output column this lane owns
  long long m = tileM + idx;               // A row this lane loads
  if (m >= nNodes) m = nNodes - 1;         // clamp loads; EXEC stays all-1s

  // C init = bias (column n is constant per lane)
  float bv = bias[n];
  v8f c;
#pragma unroll
  for (int v = 0; v < 8; ++v) c[v] = bv;

#pragma unroll
  for (int kk = 0; kk < K; kk += 32) {
    // A layout (16-bit 16x32, ISA 7.12.2): lane half h, element j:
    //   K = kk + (j/8)*16 + 8h + (j%8)  -> two contiguous float4 pairs.
    const float* arow = agg + m * K + kk + 8 * half;
    const float* xrow = xin + m * K + kk + 8 * half;
    float4 qa0 = ((const float4*)arow)[0];
    float4 qa1 = ((const float4*)arow)[1];
    float4 qa2 = ((const float4*)(arow + 16))[0];
    float4 qa3 = ((const float4*)(arow + 16))[1];
    float4 qx0 = ((const float4*)xrow)[0];
    float4 qx1 = ((const float4*)xrow)[1];
    float4 qx2 = ((const float4*)(xrow + 16))[0];
    float4 qx3 = ((const float4*)(xrow + 16))[1];
    v16bf aAgg, aX;
    pack8(aAgg, 0, qa0, qa1); pack8(aAgg, 8, qa2, qa3);
    pack8(aX,   0, qx0, qx1); pack8(aX,   8, qx2, qx3);

    // B layout (32x16): element j -> K = kk + (j/8)*16 + 8h + (j%8), col n.
    v16bf bRel, bRoot;
#pragma unroll
    for (int j = 0; j < 16; ++j) {
      int kb = kk + 8 * half + (j & 7) + 16 * (j >> 3);
      bRel[j]  = (__bf16)w_rel[kb * 64 + n];
      bRoot[j] = (__bf16)w_root[kb * 64 + n];
    }

    c = __builtin_amdgcn_wmma_f32_16x16x32_bf16(false, aAgg, false, bRel,
                                                (short)0, c, false, false);
    c = __builtin_amdgcn_wmma_f32_16x16x32_bf16(false, aX, false, bRoot,
                                                (short)0, c, false, false);
  }

  // D layout: VGPR v holds row tileM + v + 8*half, col n.
  // Uniform branch: full tiles (the common case) take the unguarded path.
  if (tileM + 16 <= nNodes) {
#pragma unroll
    for (int v = 0; v < 8; ++v) {
      float r = c[v];
      if (RELU) r = fmaxf(r, 0.0f);
      out[(tileM + v + 8 * half) * 64 + n] = r;
    }
  } else {
#pragma unroll
    for (int v = 0; v < 8; ++v) {
      long long row = tileM + v + 8 * half;
      float r = c[v];
      if (RELU) r = fmaxf(r, 0.0f);
      if (row < nNodes) out[row * 64 + n] = r;
    }
  }
}

// --------------------------- layer 3: per-node scalar dots (D_OUT == 1)
// p[i] = h2[i,:]·w3_rel ; out[i] = b3 + h2[i,:]·w3_root
__global__ void layer3_node_kernel(const float* __restrict__ h2,
                                   const float* __restrict__ w3_rel,
                                   const float* __restrict__ w3_root,
                                   const float* __restrict__ b3,
                                   float* __restrict__ p,
                                   float* __restrict__ out,
                                   long long nNodes) {
  long long i = (long long)blockIdx.x * blockDim.x + threadIdx.x;
  if (i >= nNodes) return;
  const float4* row = (const float4*)(h2 + i * 64);
  const float4* wr  = (const float4*)w3_rel;
  const float4* wo  = (const float4*)w3_root;
  float accp = 0.0f, accr = 0.0f;
#pragma unroll
  for (int k = 0; k < 16; ++k) {
    float4 v = row[k], a = wr[k], b = wo[k];
    accp += v.x * a.x + v.y * a.y + v.z * a.z + v.w * a.w;
    accr += v.x * b.x + v.y * b.y + v.z * b.z + v.w * b.w;
  }
  p[i] = accp;
  out[i] = accr + b3[0];
}

// 1 float atomic per edge (instead of 64).
__global__ void layer3_edge_kernel(const float* __restrict__ p,
                                   const int* __restrict__ src,
                                   const int* __restrict__ dst,
                                   float* __restrict__ out,
                                   long long E) {
  long long e = (long long)blockIdx.x * blockDim.x + threadIdx.x;
  if (e >= E) return;
  atom_add_f32(&out[dst[e]], p[src[e]]);
}

// ---------------------------------------------------------------- launcher
extern "C" void kernel_launch(void* const* d_in, const int* in_sizes, int n_in,
                              void* d_out, int out_size, void* d_ws, size_t ws_size,
                              hipStream_t stream) {
  const float* x       = (const float*)d_in[0];
  const float* w1_rel  = (const float*)d_in[1];
  const float* b1      = (const float*)d_in[2];
  const float* w1_root = (const float*)d_in[3];
  const float* w2_rel  = (const float*)d_in[4];
  const float* b2      = (const float*)d_in[5];
  const float* w2_root = (const float*)d_in[6];
  const float* w3_rel  = (const float*)d_in[7];
  const float* b3      = (const float*)d_in[8];
  const float* w3_root = (const float*)d_in[9];
  const int*   eidx    = (const int*)d_in[10];   // edge_index [2,E]

  const long long N = (long long)in_sizes[0] / 32;
  const long long E = (long long)in_sizes[10] / 2;
  const int* src = eidx;
  const int* dst = eidx + E;

  // workspace carve-out (N*(32+64+64+64+1)*4 bytes ≈ 90 MB for N=100k)
  float* ws   = (float*)d_ws;
  float* agg1 = ws;             // N*32
  float* h1   = agg1 + N * 32;  // N*64
  float* agg2 = h1 + N * 64;    // N*64
  float* h2   = agg2 + N * 64;  // N*64
  float* pbuf = h2 + N * 64;    // N
  float* out  = (float*)d_out;

  auto cdiv = [](long long a, long long b) -> unsigned {
    return (unsigned)((a + b - 1) / b);
  };
  const unsigned tBlocks = cdiv(N, 16);

  // ---- layer 1: agg1 = segsum(x); h1 = relu(agg1@w1_rel + x@w1_root + b1)
  zero_f32_kernel<<<cdiv(N * 32, 256), 256, 0, stream>>>(agg1, N * 32);
  scatter_add_kernel<5><<<cdiv(E * 32, 256), 256, 0, stream>>>(x, src, dst, agg1, E * 32);
  gconv_transform_kernel<32, true><<<tBlocks, 128, 0, stream>>>(
      agg1, x, w1_rel, w1_root, b1, h1, N);

  // ---- layer 2
  zero_f32_kernel<<<cdiv(N * 64, 256), 256, 0, stream>>>(agg2, N * 64);
  scatter_add_kernel<6><<<cdiv(E * 64, 256), 256, 0, stream>>>(h1, src, dst, agg2, E * 64);
  gconv_transform_kernel<64, true><<<tBlocks, 128, 0, stream>>>(
      agg2, h1, w2_rel, w2_root, b2, h2, N);

  // ---- layer 3 (scalar output): per-node dots, then 1-float/edge scatter
  layer3_node_kernel<<<cdiv(N, 256), 256, 0, stream>>>(h2, w3_rel, w3_root, b3,
                                                       pbuf, out, N);
  layer3_edge_kernel<<<cdiv(E, 256), 256, 0, stream>>>(pbuf, src, dst, out, E);
}
